// GPTNet_Lay_63136019251494
// MI455X (gfx1250) — matrix operand order repaired and checked by reading, not compiled, not run on hardware
//
#include <hip/hip_runtime.h>
#include <cstdint>

// ---------------- problem constants ----------------
#define NB  32
#define VV  25
#define TT  64
#define CC  256
#define SS  3
#define STT 2
#define CIc 64
#define KT  7
#define MROWS (NB * VV * TT)   // 51200

// ---------------- WMMA types ----------------
typedef float  v8f   __attribute__((ext_vector_type(8)));
typedef __bf16 v16bf __attribute__((ext_vector_type(16)));

union Frag16 { v16bf v; unsigned int u[8]; };

// Native fp32->bf16 (RNE) pair pack; clang lowers to v_cvt_pk_bf16_f32 on gfx1250.
__device__ __forceinline__ unsigned int pack2(float lo, float hi) {
  union { __bf16 h[2]; unsigned int u; } r;
  r.h[0] = (__bf16)lo;
  r.h[1] = (__bf16)hi;
  return r.u;
}

__device__ __forceinline__ float gelu_f(float x) {
  return 0.5f * x * (1.0f + erff(x * 0.70710678118654752440f));
}

// ---------------- WMMA GEMM: C[M,N] = A[M,K] @ B[K,N] + bias[N] ----------------
// 128x64 tile, 8 waves; wave w owns rows [16w,16w+16) x all 64 cols (4 accumulators).
#define BM2 128
#define BN2 64
#define BK2 32
#define LPA 20   // padded row stride in dwords (16 k-pairs + 4 pad); 20n mod 64 distinct for n<16

__global__ __launch_bounds__(256) void gemm_bias_kernel(
    const float* __restrict__ A, const float* __restrict__ B,
    const float* __restrict__ bias, float* __restrict__ C,
    int M, int N, int K)
{
  (void)M;
  __shared__ unsigned int lsA[BM2 * LPA];   // 10240 B
  __shared__ unsigned int lsB[BN2 * LPA];   //  5120 B
  const int tid  = threadIdx.x;
  const int row0 = blockIdx.x * BM2;
  const int col0 = blockIdx.y * BN2;
  const int lane = tid & 31;
  const int wave = tid >> 5;                // 0..7 : 16-row strip

  // A staging: thread loads 16 consecutive floats -> 8 packed dwords
  const int ar = tid >> 1;                  // 0..127
  const int ac = (tid & 1) * 8;             // dword col 0 or 8 (halfs 0..15 / 16..31)
  // B staging: thread packs k-pairs for 4 consecutive n
  const int bp  = tid >> 4;                 // k-pair 0..15
  const int bn4 = (tid & 15) * 4;           // n 0,4,...,60

  v8f acc[4] = {};

  const int m       = (lane & 15) + 16 * wave;
  const int koffA_u = (lane >> 4) * 4;      // dword offset (halfs 8*(lane>>4))
  const int nl      = lane & 15;
  const int koffB_u = (lane >> 4) * 4;      // dword offset (halfs 8*(lane>>4))

  for (int k0 = 0; k0 < K; k0 += BK2) {
    const float* ga = A + (size_t)(row0 + ar) * K + (k0 + ac * 2);
#pragma unroll
    for (int i = 0; i < 8; ++i)
      lsA[ar * LPA + ac + i] = pack2(ga[2 * i], ga[2 * i + 1]);
    const float* gb0 = B + (size_t)(k0 + 2 * bp) * N + (col0 + bn4);
    const float* gb1 = gb0 + N;
#pragma unroll
    for (int i = 0; i < 4; ++i)
      lsB[(bn4 + i) * LPA + bp] = pack2(gb0[i], gb1[i]);   // [n][k] transposed
    if (k0 + BK2 < K) {
      __builtin_prefetch(ga + BK2, 0, 3);
      __builtin_prefetch(gb0 + (size_t)BK2 * N, 0, 3);
    }
    __syncthreads();

    Frag16 fa;
    const unsigned int* pa = &lsA[m * LPA + koffA_u];
#pragma unroll
    for (int j = 0; j < 4; ++j) { fa.u[j] = pa[j]; fa.u[4 + j] = pa[8 + j]; }
#pragma unroll
    for (int nt = 0; nt < 4; ++nt) {
      Frag16 fb;
      const unsigned int* pb = &lsB[(16 * nt + nl) * LPA + koffB_u];
#pragma unroll
      for (int j = 0; j < 4; ++j) { fb.u[j] = pb[j]; fb.u[4 + j] = pb[8 + j]; }
      acc[nt] = __builtin_amdgcn_wmma_f32_16x16x32_bf16(false, fa.v, false, fb.v,
                                                        (short)0, acc[nt], false, false);
    }
    __syncthreads();
  }

  const int mb = (lane >> 4) * 8;
#pragma unroll
  for (int nt = 0; nt < 4; ++nt) {
    const int gn = col0 + 16 * nt + nl;
    const float bv = bias[gn];
#pragma unroll
    for (int i = 0; i < 8; ++i) {
      const int gm = row0 + 16 * wave + mb + i;
      C[(size_t)gm * N + gn] = acc[nt][i] + bv;
    }
  }
}

// -------- Conv over T as 7-tap shifted WMMA GEMM. Z:(N*T*V, C); W:(Cout,Cin,7) --------
__global__ __launch_bounds__(256) void conv_gemm_kernel(
    const float* __restrict__ Z, const float* __restrict__ W,
    float* __restrict__ Out)
{
  __shared__ unsigned int lsA[BM2 * LPA];
  __shared__ unsigned int lsB[BN2 * LPA];
  const int tid  = threadIdx.x;
  const int row0 = blockIdx.x * BM2;
  const int col0 = blockIdx.y * BN2;
  const int lane = tid & 31;
  const int wave = tid >> 5;

  const int ar = tid >> 1;                  // 0..127
  const int ac = (tid & 1) * 8;
  const int co = tid >> 2;                  // cout local 0..63
  const int p4 = (tid & 3) * 4;             // k-pair 0,4,8,12

  v8f acc[4] = {};

  const int m       = (lane & 15) + 16 * wave;
  const int koffA_u = (lane >> 4) * 4;
  const int nl      = lane & 15;
  const int koffB_u = (lane >> 4) * 4;

  const int gr = row0 + ar;
  const int tA = (gr / VV) % TT;            // temporal index of this A-row

  for (int tap = 0; tap < KT; ++tap) {
    const int dt = tap - (KT - 1) / 2;
    const int ts = tA + dt;
    const bool valid = (ts >= 0) && (ts < TT);
    for (int kc = 0; kc < CC; kc += BK2) {
      const float* ga = Z + (size_t)(gr + dt * VV) * CC + (kc + ac * 2);
#pragma unroll
      for (int i = 0; i < 8; ++i)
        lsA[ar * LPA + ac + i] = valid ? pack2(ga[2 * i], ga[2 * i + 1]) : 0u;
      const float* gw = W + (size_t)(col0 + co) * (CC * KT) + (size_t)kc * KT + tap;
#pragma unroll
      for (int i = 0; i < 4; ++i) {
        const int p = p4 + i;
        lsB[co * LPA + p] = pack2(gw[(size_t)(2 * p) * KT], gw[(size_t)(2 * p + 1) * KT]);
      }
      __syncthreads();

      Frag16 fa;
      const unsigned int* pa = &lsA[m * LPA + koffA_u];
#pragma unroll
      for (int j = 0; j < 4; ++j) { fa.u[j] = pa[j]; fa.u[4 + j] = pa[8 + j]; }
#pragma unroll
      for (int nt = 0; nt < 4; ++nt) {
        Frag16 fb;
        const unsigned int* pb = &lsB[(16 * nt + nl) * LPA + koffB_u];
#pragma unroll
        for (int j = 0; j < 4; ++j) { fb.u[j] = pb[j]; fb.u[4 + j] = pb[8 + j]; }
        acc[nt] = __builtin_amdgcn_wmma_f32_16x16x32_bf16(false, fa.v, false, fb.v,
                                                          (short)0, acc[nt], false, false);
      }
      __syncthreads();
    }
  }

  const int mb = (lane >> 4) * 8;
#pragma unroll
  for (int nt = 0; nt < 4; ++nt) {
    const int gn = col0 + 16 * nt + nl;
#pragma unroll
    for (int i = 0; i < 8; ++i) {
      const int gm = row0 + 16 * wave + mb + i;
      Out[(size_t)gm * CC + gn] = acc[nt][i];
    }
  }
}

// -------- spatial attention: att[n,u,v,s] = tanh(<q,k>/(CI*T))*alpha[s] + att0[u,v,s] --------
__global__ __launch_bounds__(256) void spatial_att_kernel(
    const float* __restrict__ qk, const float* __restrict__ alphas,
    const float* __restrict__ att0, float* __restrict__ att)
{
  const int b = blockIdx.x;                     // (n*V + u)*V + v
  const int v = b % VV;
  const int u = (b / VV) % VV;
  const int n = b / (VV * VV);
  const size_t qbase = (size_t)(n * VV + u) * TT * CIc * (2 * SS);
  const size_t kbase = (size_t)(n * VV + v) * TT * CIc * (2 * SS);
  float acc0 = 0.f, acc1 = 0.f, acc2 = 0.f;
  for (int i = threadIdx.x; i < TT * CIc; i += 256) {
    const size_t o = (size_t)i * (2 * SS);
    const float* qp = qk + qbase + o;
    const float* kp = qk + kbase + o + SS;
    acc0 += qp[0] * kp[0];
    acc1 += qp[1] * kp[1];
    acc2 += qp[2] * kp[2];
  }
  __shared__ float red[3][256];
  red[0][threadIdx.x] = acc0; red[1][threadIdx.x] = acc1; red[2][threadIdx.x] = acc2;
  __syncthreads();
  for (int off = 128; off > 0; off >>= 1) {
    if (threadIdx.x < off) {
      red[0][threadIdx.x] += red[0][threadIdx.x + off];
      red[1][threadIdx.x] += red[1][threadIdx.x + off];
      red[2][threadIdx.x] += red[2][threadIdx.x + off];
    }
    __syncthreads();
  }
  if (threadIdx.x < SS) {
    const int s = threadIdx.x;
    const float d = tanhf(red[s][0] * (1.0f / (float)(CIc * TT))) * alphas[s]
                  + att0[((size_t)u * VV + v) * SS + s];
    att[(size_t)b * SS + s] = d;
  }
}

// -------- spatial mix: y[n,v,t,c*S+s] = sum_u x[n,u,t,c] * att[n,u,v,s] --------
__global__ __launch_bounds__(256) void spatial_mix_kernel(
    const float* __restrict__ x, const float* __restrict__ att, float* __restrict__ y)
{
  const size_t idx = (size_t)blockIdx.x * 256 + threadIdx.x;
  const int s = (int)(idx % SS); size_t r = idx / SS;
  const int c = (int)(r % CC); r /= CC;
  const int t = (int)(r % TT); r /= TT;
  const int v = (int)(r % VV);
  const int n = (int)(r / VV);
  float acc = 0.f;
#pragma unroll 5
  for (int u = 0; u < VV; ++u)
    acc += x[((size_t)(n * VV + u) * TT + t) * CC + c]
         * att[((size_t)(n * VV + u) * VV + v) * SS + s];
  y[idx] = acc;
}

// -------- LayerNorm over `inner` + residual + GELU --------
__global__ __launch_bounds__(256) void ln_res_gelu_kernel(
    const float* __restrict__ res, const float* __restrict__ tin,
    const float* __restrict__ g, const float* __restrict__ b,
    float* __restrict__ out, int inner)
{
  __shared__ float s1[256], s2[256];
  const size_t base = (size_t)blockIdx.x * inner;
  float s = 0.f, q = 0.f;
  for (int i = threadIdx.x; i < inner; i += 256) {
    const float v = tin[base + i];
    s += v; q += v * v;
  }
  s1[threadIdx.x] = s; s2[threadIdx.x] = q;
  __syncthreads();
  for (int off = 128; off > 0; off >>= 1) {
    if (threadIdx.x < off) {
      s1[threadIdx.x] += s1[threadIdx.x + off];
      s2[threadIdx.x] += s2[threadIdx.x + off];
    }
    __syncthreads();
  }
  const float mean = s1[0] / (float)inner;
  const float var  = s2[0] / (float)inner - mean * mean;
  const float rstd = rsqrtf(var + 1e-5f);
  for (int i = threadIdx.x; i < inner; i += 256) {
    const float v = (tin[base + i] - mean) * rstd * g[i] + b[i];
    out[base + i] = gelu_f(res[base + i] + v);
  }
}

// -------- transpose (N,V,T,C) -> (N,T,V,C) --------
__global__ __launch_bounds__(256) void transpose_vt_kernel(
    const float* __restrict__ in, float* __restrict__ out)
{
  const size_t idx = (size_t)blockIdx.x * 256 + threadIdx.x;
  const int c = (int)(idx & (CC - 1)); size_t r = idx >> 8;
  const int t = (int)(r % TT); r /= TT;
  const int v = (int)(r % VV);
  const int n = (int)(r / VV);
  out[((size_t)(n * TT + t) * VV + v) * CC + c] = in[idx];
}

// -------- temporal attention logits (masked, f & b) --------
__global__ __launch_bounds__(256) void temporal_att_kernel(
    const float* __restrict__ qkt, const float* __restrict__ alf,
    const float* __restrict__ alb, float* __restrict__ attf, float* __restrict__ attb)
{
  const int J = 4 * STT;                        // 8
  const int bidx = blockIdx.x;                  // (n*T + t)*T + q
  const int q = bidx % TT;
  const int t = (bidx / TT) % TT;
  const int n = bidx / (TT * TT);
  const size_t tb = (size_t)(n * TT + t) * VV * CIc * J;
  const size_t qb = (size_t)(n * TT + q) * VV * CIc * J;
  float a0 = 0.f, a1 = 0.f, a2 = 0.f, a3 = 0.f;
  for (int i = threadIdx.x; i < VV * CIc; i += 256) {
    const size_t o = (size_t)i * J;
    const float* pt = qkt + tb + o;
    const float* pq = qkt + qb + o;
    a0 += pt[0] * pq[4];   // q_f[s=0] * k_f[s=0]
    a1 += pt[1] * pq[5];   // q_f[s=1] * k_f[s=1]
    a2 += pt[2] * pq[6];   // q_b[s=0] * k_b[s=0]
    a3 += pt[3] * pq[7];   // q_b[s=1] * k_b[s=1]
  }
  __shared__ float red[4][256];
  red[0][threadIdx.x] = a0; red[1][threadIdx.x] = a1;
  red[2][threadIdx.x] = a2; red[3][threadIdx.x] = a3;
  __syncthreads();
  for (int off = 128; off > 0; off >>= 1) {
    if (threadIdx.x < off) {
      red[0][threadIdx.x] += red[0][threadIdx.x + off];
      red[1][threadIdx.x] += red[1][threadIdx.x + off];
      red[2][threadIdx.x] += red[2][threadIdx.x + off];
      red[3][threadIdx.x] += red[3][threadIdx.x + off];
    }
    __syncthreads();
  }
  if (threadIdx.x < 4) {
    const int s = threadIdx.x & 1;
    const bool isb = threadIdx.x >= 2;
    float val = tanhf(red[threadIdx.x][0] * (1.0f / (float)(CIc * VV)));
    if (!isb) {
      val *= alf[s] * ((t >= q) ? 1.f : 0.f);   // bmask.T
      attf[(size_t)bidx * STT + s] = val;
    } else {
      val *= alb[s] * ((q >= t) ? 1.f : 0.f);   // bmask
      attb[(size_t)bidx * STT + s] = val;
    }
  }
}

// -------- temporal mix: z[n,q,v, fb*512 + c*ST+s] = sum_t y[n,t,v,c]*att[n,t,q,s] --------
__global__ __launch_bounds__(256) void temporal_mix_kernel(
    const float* __restrict__ y, const float* __restrict__ attf,
    const float* __restrict__ attb, float* __restrict__ z)
{
  const size_t idx = (size_t)blockIdx.x * 256 + threadIdx.x;
  const int off = (int)(idx & 1023); size_t r = idx >> 10;
  const int v = (int)(r % VV); r /= VV;
  const int q = (int)(r % TT);
  const int n = (int)(r / TT);
  const int fb = off >> 9;
  const int cs = off & 511;
  const int s = cs % STT;
  const int c = cs / STT;
  const float* att = fb ? attb : attf;
  float acc = 0.f;
#pragma unroll 8
  for (int t = 0; t < TT; ++t)
    acc += y[((size_t)(n * TT + t) * VV + v) * CC + c]
         * att[((size_t)(n * TT + t) * TT + q) * STT + s];
  z[idx] = acc;
}

// -------- conv bias + BN + residual + GELU + output transpose to (N,V,T,C) --------
__global__ __launch_bounds__(256) void final_kernel(
    const float* __restrict__ z, const float* __restrict__ conv,
    const float* __restrict__ cb, const float* __restrict__ gg,
    const float* __restrict__ gb, const float* __restrict__ gm,
    const float* __restrict__ gv, float* __restrict__ out)
{
  const size_t idx = (size_t)blockIdx.x * 256 + threadIdx.x;  // (n,t,v,c)
  const int c = (int)(idx & (CC - 1)); size_t r = idx >> 8;
  const int v = (int)(r % VV); r /= VV;
  const int t = (int)(r % TT);
  const int n = (int)(r / TT);
  float val = conv[idx] + cb[c];
  val = (val - gm[c]) * rsqrtf(gv[c] + 1e-5f) * gg[c] + gb[c];
  out[((size_t)(n * VV + v) * TT + t) * CC + c] = gelu_f(z[idx] + val);
}

// ---------------- host orchestration ----------------
extern "C" void kernel_launch(void* const* d_in, const int* in_sizes, int n_in,
                              void* d_out, int out_size, void* d_ws, size_t ws_size,
                              hipStream_t stream) {
  (void)in_sizes; (void)n_in; (void)out_size; (void)ws_size;
  const float* x      = (const float*)d_in[0];
  const float* Wqk_s  = (const float*)d_in[1];
  const float* bqk_s  = (const float*)d_in[2];
  const float* alphas = (const float*)d_in[3];
  const float* att0   = (const float*)d_in[4];
  const float* Wo_s   = (const float*)d_in[5];
  const float* bo_s   = (const float*)d_in[6];
  const float* g_os   = (const float*)d_in[7];
  const float* b_os   = (const float*)d_in[8];
  const float* Wff_s  = (const float*)d_in[9];
  const float* bff_s  = (const float*)d_in[10];
  const float* g_ffs  = (const float*)d_in[11];
  const float* b_ffs  = (const float*)d_in[12];
  const float* Wqk_t  = (const float*)d_in[13];
  const float* bqk_t  = (const float*)d_in[14];
  const float* alf    = (const float*)d_in[15];
  const float* alb    = (const float*)d_in[16];
  const float* Wo_t   = (const float*)d_in[17];
  const float* bo_t   = (const float*)d_in[18];
  const float* g_ot   = (const float*)d_in[19];
  const float* b_ot   = (const float*)d_in[20];
  const float* Wff_t  = (const float*)d_in[21];
  const float* bff_t  = (const float*)d_in[22];
  const float* g_fft  = (const float*)d_in[23];
  const float* b_fft  = (const float*)d_in[24];
  const float* conv_w = (const float*)d_in[25];
  const float* conv_b = (const float*)d_in[26];
  const float* bn_g   = (const float*)d_in[27];
  const float* bn_b   = (const float*)d_in[28];
  const float* bn_m   = (const float*)d_in[29];
  const float* bn_v   = (const float*)d_in[30];
  float* out = (float*)d_out;

  // workspace layout (floats)
  float* big  = (float*)d_ws;                         // 51200*1024 (qk/qkt/y_big/z_big)
  float* tb   = big  + (size_t)MROWS * 1024;          // 51200*256
  float* ya   = tb   + (size_t)MROWS * CC;            // 51200*256
  float* yb   = ya   + (size_t)MROWS * CC;            // 51200*256
  float* yt   = yb   + (size_t)MROWS * CC;            // 51200*256
  float* atts = yt   + (size_t)MROWS * CC;            // 60000 (padded 65536)
  float* attf = atts + 65536;                         // 262144
  float* attb = attf + 262144;                        // 262144

  const dim3 blk(256);
  const int MT = MROWS / BM2;                         // 400

  // --- spatial block ---
  gemm_bias_kernel<<<dim3(MT, 384 / 64), blk, 0, stream>>>(x, Wqk_s, bqk_s, big, MROWS, 384, CC);
  spatial_att_kernel<<<dim3(NB * VV * VV), blk, 0, stream>>>(big, alphas, att0, atts);
  spatial_mix_kernel<<<dim3((NB * VV * TT * CC * SS) / 256), blk, 0, stream>>>(x, atts, big);
  gemm_bias_kernel<<<dim3(MT, CC / 64), blk, 0, stream>>>(big, Wo_s, bo_s, tb, MROWS, CC, SS * CC);
  ln_res_gelu_kernel<<<dim3(NB * VV), blk, 0, stream>>>(x, tb, g_os, b_os, ya, TT * CC);
  gemm_bias_kernel<<<dim3(MT, CC / 64), blk, 0, stream>>>(ya, Wff_s, bff_s, tb, MROWS, CC, CC);
  ln_res_gelu_kernel<<<dim3(NB * VV), blk, 0, stream>>>(ya, tb, g_ffs, b_ffs, yb, TT * CC);

  // --- temporal block ---
  transpose_vt_kernel<<<dim3((NB * VV * TT * CC) / 256), blk, 0, stream>>>(yb, yt);
  gemm_bias_kernel<<<dim3(MT, 512 / 64), blk, 0, stream>>>(yt, Wqk_t, bqk_t, big, MROWS, 512, CC);
  temporal_att_kernel<<<dim3(NB * TT * TT), blk, 0, stream>>>(big, alf, alb, attf, attb);
  temporal_mix_kernel<<<dim3((NB * TT * VV * 1024) / 256), blk, 0, stream>>>(yt, attf, attb, big);
  gemm_bias_kernel<<<dim3(MT, CC / 64), blk, 0, stream>>>(big, Wo_t, bo_t, tb, MROWS, CC, 2 * STT * CC);
  ln_res_gelu_kernel<<<dim3(NB * TT), blk, 0, stream>>>(yt, tb, g_ot, b_ot, ya, VV * CC);
  gemm_bias_kernel<<<dim3(MT, CC / 64), blk, 0, stream>>>(ya, Wff_t, bff_t, tb, MROWS, CC, CC);
  ln_res_gelu_kernel<<<dim3(NB * TT), blk, 0, stream>>>(ya, tb, g_fft, b_fft, yb, VV * CC);

  // --- conv + BN + output ---
  conv_gemm_kernel<<<dim3(MT, CC / 64), blk, 0, stream>>>(yb, conv_w, tb);
  final_kernel<<<dim3((NB * TT * VV * CC) / 256), blk, 0, stream>>>(yb, tb, conv_b, bn_g, bn_b, bn_m, bn_v, out);
}
